// SDPA_57432302682406
// MI455X (gfx1250) — matrix operand order repaired and checked by reading, not compiled, not run on hardware
//
#include <hip/hip_runtime.h>

// ---------------------------------------------------------------------------
// Causal GQA flash-attention forward for MI455X (gfx1250, wave32, WMMA).
// q: [2,32,2048,128] f32, k/v: [2,8,2048,128] f32, out: [2,32,2048,128] f32.
// bf16 WMMA (v_wmma_f32_16x16x32_bf16) for QK^T and PV, f32 accumulation,
// online softmax with DPP16 butterfly reductions (no LDS traffic),
// global_prefetch of the next K/V tile. 8 waves/block, 128 q-rows per block.
// ---------------------------------------------------------------------------

typedef __attribute__((ext_vector_type(16))) __bf16 bf16x16;
typedef __attribute__((ext_vector_type(8)))  __bf16 bf16x8;
typedef __attribute__((ext_vector_type(8)))  float  f32x8;

#define SEQ   2048
#define DH    128
#define HQN   32
#define HKN   8
#define BATCH 2
#define BR    128            // q rows per block (8 waves * 16)
#define BC    64             // k rows per tile
#define NWAVE 8
#define KSTRIDE (DH + 8)     // 136 elems -> 272B rows (16B aligned)
#define VSTRIDE (BC + 8)     // 72 elems  -> 144B rows (16B aligned)

__device__ __forceinline__ f32x8 vzero8() {
    f32x8 z;
#pragma unroll
    for (int i = 0; i < 8; ++i) z[i] = 0.0f;
    return z;
}

__device__ __forceinline__ bf16x16 combine16(bf16x8 lo, bf16x8 hi) {
    bf16x16 r;
#pragma unroll
    for (int i = 0; i < 8; ++i) { r[i] = lo[i]; r[8 + i] = hi[i]; }
    return r;
}

__device__ __forceinline__ f32x8 wmma_bf16(bf16x16 a, bf16x16 b, f32x8 c) {
    return __builtin_amdgcn_wmma_f32_16x16x32_bf16(
        /*neg_a=*/false, a, /*neg_b=*/false, b,
        /*c_mod=*/(short)0, c, /*reuse_a=*/false, /*reuse_b=*/false);
}

// ---- 16-lane butterfly exchange via DPP16 (VALU only, no LDS) --------------
// Stage controls: quad_perm(1,0,3,2)=0xB1 (xor1), quad_perm(2,3,0,1)=0x4E
// (xor2), row_half_mirror=0x141 (== xor4 once quad-uniform), row_mirror=0x140
// (== xor8 once half-uniform). Result is broadcast to all 16 lanes of a half.
#if __has_builtin(__builtin_amdgcn_update_dpp)
template <int CTRL>
__device__ __forceinline__ float dpp_bfly(float x) {
    const int i = __builtin_bit_cast(int, x);
    const int r = __builtin_amdgcn_update_dpp(0, i, CTRL, 0xF, 0xF, true);
    return __builtin_bit_cast(float, r);
}
__device__ __forceinline__ float row16_allmax(float v) {
    v = fmaxf(v, dpp_bfly<0xB1>(v));
    v = fmaxf(v, dpp_bfly<0x4E>(v));
    v = fmaxf(v, dpp_bfly<0x141>(v));
    v = fmaxf(v, dpp_bfly<0x140>(v));
    return v;
}
__device__ __forceinline__ float row16_allsum(float v) {
    v += dpp_bfly<0xB1>(v);
    v += dpp_bfly<0x4E>(v);
    v += dpp_bfly<0x141>(v);
    v += dpp_bfly<0x140>(v);
    return v;
}
#else
__device__ __forceinline__ float row16_allmax(float v) {
#pragma unroll
    for (int off = 1; off < 16; off <<= 1) v = fmaxf(v, __shfl_xor(v, off, 32));
    return v;
}
__device__ __forceinline__ float row16_allsum(float v) {
#pragma unroll
    for (int off = 1; off < 16; off <<= 1) v += __shfl_xor(v, off, 32);
    return v;
}
#endif

__global__ __launch_bounds__(256, 1) void fa_fwd_gqa_causal(
    const float* __restrict__ q, const float* __restrict__ k,
    const float* __restrict__ v, float* __restrict__ out)
{
    __shared__ __bf16 kbuf[BC * KSTRIDE];      // [k_row][d]        bf16
    __shared__ __bf16 vtbuf[DH * VSTRIDE];     // [d][k_row]        bf16 (transposed)
    __shared__ __bf16 pbuf[NWAVE * 16 * BC];   // per-wave P tile   bf16

    const int tid  = threadIdx.x;
    const int wave = tid >> 5;
    const int lane = tid & 31;
    const int half = lane >> 4;    // which 16-lane half of the wave
    const int l16  = lane & 15;

    const int qb = blockIdx.x;
    const int h  = blockIdx.y;
    const int b  = blockIdx.z;
    const int hk = h / (HQN / HKN);            // GQA: 4 q-heads share a kv-head

    const int   q0    = qb * BR;
    const int   qrow  = q0 + wave * 16 + l16;  // A-layout: row M = lane%16
    const float scale = 0.08838834764831845f;  // 1/sqrt(128)

    const float* kbase_p = k + (((size_t)b * HKN + hk) * SEQ) * DH;
    const float* vbase_p = v + (((size_t)b * HKN + hk) * SEQ) * DH;

    // ---- Q fragments: 16x128 per wave as 4 A-frags (16x32 bf16), pre-scaled.
    // A layout (16-bit 16x32): element e of lane maps to K = e + 8*((e>>3)+half).
    bf16x16 qa[4];
    {
        const float* qp = q + (((size_t)b * HQN + h) * SEQ + qrow) * DH;
#pragma unroll
        for (int c = 0; c < 4; ++c) {
            const int d0 = 32 * c + 8 * half;        // e = 0..7
            const int d1 = 32 * c + 16 + 8 * half;   // e = 8..15
            float tmp[16];
            *(float4*)(tmp + 0)  = *(const float4*)(qp + d0);
            *(float4*)(tmp + 4)  = *(const float4*)(qp + d0 + 4);
            *(float4*)(tmp + 8)  = *(const float4*)(qp + d1);
            *(float4*)(tmp + 12) = *(const float4*)(qp + d1 + 4);
#pragma unroll
            for (int e = 0; e < 16; ++e) qa[c][e] = (__bf16)(tmp[e] * scale);
        }
    }

    // ---- flash accumulators (C layout: VGPR r, row M = r + 8*half, col N = lane%16)
    f32x8 acc[8];
#pragma unroll
    for (int i = 0; i < 8; ++i) acc[i] = vzero8();
    f32x8 m_run, l_run;
#pragma unroll
    for (int r = 0; r < 8; ++r) { m_run[r] = -INFINITY; l_run[r] = 0.0f; }

    const int nkt = (q0 + BR) / BC;            // causal: only tiles up to diagonal

    for (int kt = 0; kt < nkt; ++kt) {
        // ---- cooperative K/V tile load: f32 -> bf16, V transposed in LDS ----
        {
            const float* kp = kbase_p + (size_t)kt * BC * DH;
            const float* vp = vbase_p + (size_t)kt * BC * DH;
#pragma unroll
            for (int i = 0; i < 8; ++i) {
                const int idx = tid + i * 256;         // 2048 float4 slots
                const int row = idx >> 5;              // 32 float4 per row
                const int c4  = (idx & 31) * 4;
                const float4 kf = *(const float4*)(kp + row * DH + c4);
                const float4 vf = *(const float4*)(vp + row * DH + c4);
                __bf16* kd = &kbuf[row * KSTRIDE + c4];
                kd[0] = (__bf16)kf.x; kd[1] = (__bf16)kf.y;
                kd[2] = (__bf16)kf.z; kd[3] = (__bf16)kf.w;
                vtbuf[(c4 + 0) * VSTRIDE + row] = (__bf16)vf.x;
                vtbuf[(c4 + 1) * VSTRIDE + row] = (__bf16)vf.y;
                vtbuf[(c4 + 2) * VSTRIDE + row] = (__bf16)vf.z;
                vtbuf[(c4 + 3) * VSTRIDE + row] = (__bf16)vf.w;
            }
        }
        __syncthreads();

        // ---- prefetch next K/V tile (one 128B line per thread covers 32KB) --
        if (kt + 1 < nkt) {
            const float* kpn = kbase_p + (size_t)(kt + 1) * BC * DH + tid * 32;
            const float* vpn = vbase_p + (size_t)(kt + 1) * BC * DH + tid * 32;
            __builtin_prefetch(kpn, 0, 1);
            __builtin_prefetch(vpn, 0, 1);
        }

        // ---- S = (Q*scale) @ K^T : 4 N-tiles x 4 K-steps of WMMA ----
        f32x8 s[4];
#pragma unroll
        for (int nt = 0; nt < 4; ++nt) {
            s[nt] = vzero8();
#pragma unroll
            for (int c = 0; c < 4; ++c) {
                // B layout (32x16): lane col N = lane%16, element e -> K = e + 8*((e>>3)+half)
                const __bf16* bp = &kbuf[(nt * 16 + l16) * KSTRIDE + 32 * c + 8 * half];
                const bf16x8 blo = *(const bf16x8*)(bp);
                const bf16x8 bhi = *(const bf16x8*)(bp + 16);
                s[nt] = wmma_bf16(qa[c], combine16(blo, bhi), s[nt]);
            }
        }

        // ---- causal mask (only tiles crossing the diagonal) ----
        const int kbase = kt * BC;
        if (kbase + BC - 1 > q0 + wave * 16) {
#pragma unroll
            for (int nt = 0; nt < 4; ++nt)
#pragma unroll
                for (int r = 0; r < 8; ++r) {
                    const int qg = q0 + wave * 16 + half * 8 + r;
                    const int kg = kbase + nt * 16 + l16;
                    if (kg > qg) s[nt][r] = -INFINITY;
                }
        }

        // ---- online softmax (rows spread across 16 lanes of a half) ----
        f32x8 escale;
#pragma unroll
        for (int r = 0; r < 8; ++r) {
            float rm = s[0][r];
#pragma unroll
            for (int nt = 1; nt < 4; ++nt) rm = fmaxf(rm, s[nt][r]);
            rm = row16_allmax(rm);
            const float mn = fmaxf(m_run[r], rm);
            escale[r] = __expf(m_run[r] - mn);
            m_run[r]  = mn;
        }

        f32x8 rsum = vzero8();
#pragma unroll
        for (int nt = 0; nt < 4; ++nt)
#pragma unroll
            for (int r = 0; r < 8; ++r) {
                const float p = __expf(s[nt][r] - m_run[r]);
                s[nt][r] = p;
                rsum[r] += p;
            }
#pragma unroll
        for (int r = 0; r < 8; ++r) {
            const float rs = row16_allsum(rsum[r]);
            l_run[r] = l_run[r] * escale[r] + rs;
        }
#pragma unroll
        for (int nt = 0; nt < 8; ++nt)
#pragma unroll
            for (int r = 0; r < 8; ++r) acc[nt][r] *= escale[r];

        // ---- P: C layout -> LDS -> A layout (per-wave scratch, no barrier) ----
        __bf16* pw = &pbuf[wave * 16 * BC];
#pragma unroll
        for (int nt = 0; nt < 4; ++nt)
#pragma unroll
            for (int r = 0; r < 8; ++r)
                pw[(r + 8 * half) * BC + nt * 16 + l16] = (__bf16)s[nt][r];

        bf16x16 pa[2];
#pragma unroll
        for (int c2 = 0; c2 < 2; ++c2) {
            const __bf16* ap = pw + l16 * BC + 32 * c2 + 8 * half;
            const bf16x8 alo = *(const bf16x8*)(ap);       // K = e       (+8*half)
            const bf16x8 ahi = *(const bf16x8*)(ap + 16);  // K = e+8     (+8*half)
            pa[c2] = combine16(alo, ahi);
        }

        // ---- O += P @ V : 8 N-tiles (D=128) x 2 K-steps of WMMA ----
#pragma unroll
        for (int nt = 0; nt < 8; ++nt)
#pragma unroll
            for (int c2 = 0; c2 < 2; ++c2) {
                const __bf16* bp = &vtbuf[(nt * 16 + l16) * VSTRIDE + 32 * c2 + 8 * half];
                const bf16x8 blo = *(const bf16x8*)(bp);
                const bf16x8 bhi = *(const bf16x8*)(bp + 16);
                acc[nt] = wmma_bf16(pa[c2], combine16(blo, bhi), acc[nt]);
            }

        __syncthreads();   // protect kbuf/vtbuf before next tile overwrites
    }

    // ---- epilogue: O = acc / l, store f32 ----
    f32x8 rinv;
#pragma unroll
    for (int r = 0; r < 8; ++r) rinv[r] = 1.0f / l_run[r];

    float* op = out + (((size_t)b * HQN + h) * SEQ + q0 + wave * 16) * DH;
#pragma unroll
    for (int nt = 0; nt < 8; ++nt)
#pragma unroll
        for (int r = 0; r < 8; ++r)
            op[(r + 8 * half) * DH + nt * 16 + l16] = acc[nt][r] * rinv[r];
}

extern "C" void kernel_launch(void* const* d_in, const int* in_sizes, int n_in,
                              void* d_out, int out_size, void* d_ws, size_t ws_size,
                              hipStream_t stream) {
    (void)in_sizes; (void)n_in; (void)out_size; (void)d_ws; (void)ws_size;
    const float* q = (const float*)d_in[0];
    const float* k = (const float*)d_in[1];
    const float* v = (const float*)d_in[2];
    float* out = (float*)d_out;

    dim3 grid(SEQ / BR, HQN, BATCH);   // (16, 32, 2)
    fa_fwd_gqa_causal<<<grid, 256, 0, stream>>>(q, k, v, out);
}
